// CRF_11364483465320
// MI455X (gfx1250) — matrix operand order repaired and checked by reading, not compiled, not run on hardware
//
#include <hip/hip_runtime.h>

// CRF on MI455X (gfx1250, wave32).
// Phase 1 (crf_main, 76 blocks x 512):
//   blocks 0-3  : forward algorithm. logsumexp-plus GEMM factored exactly as
//                 S = exp(alpha - rowmax) @ exp(trans - M)  -> v_wmma_f32_16x16x32_f16
//                 2 barriers/step, emissions software-pipelined 1 step ahead.
//   blocks 4-67 : Viterbi max-plus scan, trans column in 32 VGPRs/thread,
//                 emissions prefetched 1 step ahead, backpointers (u8) written
//                 into d_out's onehot region (scratch).
//   blocks 68-75: gold path score reduction.
// Phase 2 (crf_backtrace): per-batch pointer chase through LDS-staged bp chunks
//                 (+ global_prefetch of the next chunk), writes nll and path.
// Phase 3 (crf_onehot): massively parallel float4 one-hot fill (overwrites bp scratch).

#define B_   64
#define T_   2048
#define C_   128
#define TM1  2047

typedef _Float16 v16h  __attribute__((ext_vector_type(16)));
typedef _Float16 f16x8 __attribute__((ext_vector_type(8)));
typedef _Float16 f16x4 __attribute__((ext_vector_type(4)));
typedef float    v8f   __attribute__((ext_vector_type(8)));

// d_ws layout (tiny: ~132KB)
#define WS_LOGZ 0        // 64 floats
#define WS_GOLD 256      // 64 floats
#define WS_BEST 512      // 64 ints
#define WS_PATH 1024     // 64*2048 u8 decoded paths

struct PV { float v; int i; };   // packed partial (value, argmax index)

struct FwdShared {
  alignas(16) float     alpha[16 * 128];   // 8 KB  current alpha (f32)
  alignas(16) float     emit[16 * 128];    // 8 KB  staged emissions(t)
  alignas(16) _Float16  P[16 * 128];       // 4 KB  exp(alpha - rowmax) f16
  alignas(16) _Float16  E[128 * 128];      // 32 KB exp(trans - M) f16 (constant)
  float m[16];
  float wred[16];
};
struct VitShared {
  alignas(16) float v[128];
  alignas(8)  PV    pvpi[512];
};
struct GoldShared {
  float red[512];
};
union SMem { FwdShared f; VitShared v; GoldShared g; };   // ~53 KB < per-WG LDS

__global__ __launch_bounds__(512)
void crf_main(const float* __restrict__ emissions,
              const int*   __restrict__ tags,
              const float* __restrict__ trans,
              const float* __restrict__ start,
              const float* __restrict__ end_,
              float*       __restrict__ out,
              unsigned char* __restrict__ ws)
{
  __shared__ SMem sm;
  const int tid = threadIdx.x;
  const int blk = blockIdx.x;
  unsigned char* bp   = (unsigned char*)out + 256;  // backptr scratch inside onehot region
  float* logZ = (float*)(ws + WS_LOGZ);
  float* gold = (float*)(ws + WS_GOLD);
  int*   bestL = (int*)(ws + WS_BEST);

  if (blk < 4) {
    // ----------------- forward (log-partition) via WMMA -----------------
    const int b0 = blk * 16;
    const int wave = tid >> 5, lane = tid & 31;
    const int hi = lane >> 4, l15 = lane & 15;
    const int n0 = wave * 16;              // N-tile for waves 0..7
    const int r  = wave;                   // row owned end-to-end in phase A
    const int c4 = lane * 4;               // 4 contiguous columns per lane

    // global transition max M
    float lm = -3.4e38f;
    for (int idx = tid; idx < C_ * C_; idx += 512) lm = fmaxf(lm, trans[idx]);
    #pragma unroll
    for (int off = 16; off >= 1; off >>= 1) lm = fmaxf(lm, __shfl_xor(lm, off));
    if (lane == 0) sm.f.wred[wave] = lm;
    __syncthreads();
    if (tid == 0) {
      float mm = sm.f.wred[0];
      #pragma unroll
      for (int w = 1; w < 16; ++w) mm = fmaxf(mm, sm.f.wred[w]);
      sm.f.m[0] = mm;
    }
    __syncthreads();
    const float MT = sm.f.m[0];
    __syncthreads();

    // E = exp(trans - M) in f16, resident in LDS for all 2047 steps
    for (int idx = tid; idx < C_ * C_; idx += 512)
      sm.f.E[idx] = (_Float16)__expf(trans[idx] - MT);

    // alpha0 = start + emissions[:,0]; preload emissions(t=1) into registers
    const size_t rowBase = (size_t)(b0 + r) * T_ * C_;
    {
      float4 e0 = *(const float4*)&emissions[rowBase + c4];
      float4 st = *(const float4*)&start[c4];
      float4 a0; a0.x = e0.x + st.x; a0.y = e0.y + st.y;
                 a0.z = e0.z + st.z; a0.w = e0.w + st.w;
      *(float4*)&sm.f.alpha[r * 128 + c4] = a0;
    }
    float4 er = *(const float4*)&emissions[rowBase + C_ + c4];   // t = 1
    __syncthreads();

    for (int t = 1; t < T_; ++t) {
      // phase A (all 16 waves, wave r owns row r):
      //  stage emissions(t), row max, P = exp(alpha - m), prefetch emissions(t+1)
      *(float4*)&sm.f.emit[r * 128 + c4] = er;
      float4 av = *(const float4*)&sm.f.alpha[r * 128 + c4];
      float mx = fmaxf(fmaxf(av.x, av.y), fmaxf(av.z, av.w));
      #pragma unroll
      for (int off = 16; off >= 1; off >>= 1) mx = fmaxf(mx, __shfl_xor(mx, off));
      if (lane == 0) sm.f.m[r] = mx;
      f16x4 pv;
      pv[0] = (_Float16)__expf(av.x - mx);
      pv[1] = (_Float16)__expf(av.y - mx);
      pv[2] = (_Float16)__expf(av.z - mx);
      pv[3] = (_Float16)__expf(av.w - mx);
      *(f16x4*)&sm.f.P[r * 128 + c4] = pv;
      if (t + 1 < T_)
        er = *(const float4*)&emissions[rowBase + (size_t)(t + 1) * C_ + c4];
      __syncthreads();

      // phase B: S = P @ E : waves 0..7, one 16x16 tile each, K=128 (4 x wmma k32)
      if (tid < 256) {
        v8f acc = {};
        #pragma unroll
        for (int kb = 0; kb < 4; ++kb) {
          // A fragment (ISA 16-bit A 16x32 layout): row=l15, K = 8*hi+{0..7, 16..23}
          const _Float16* pa = &sm.f.P[l15 * 128 + kb * 32 + hi * 8];
          f16x8 a_lo = *(const f16x8*)pa;
          f16x8 a_hi = *(const f16x8*)(pa + 16);
          v16h A = __builtin_shufflevector(a_lo, a_hi,
                     0,1,2,3,4,5,6,7,8,9,10,11,12,13,14,15);
          // B fragment: col=n0+l15, K = 8*hi+{0..7, 16..23} (same as A)
          const _Float16* pb = &sm.f.E[(kb * 32 + hi * 8) * 128 + n0 + l15];
          v16h Bf;
          #pragma unroll
          for (int s = 0; s < 16; ++s)
            Bf[s] = pb[((s < 8) ? s : s + 8) * 128];
          acc = __builtin_amdgcn_wmma_f32_16x16x32_f16(
                  false, A, false, Bf, (short)0, acc, false, false);
        }
        // alpha_new = m + M + log(S) + emit   (D layout: VGPR g -> M=g+8*hi, N=l15)
        #pragma unroll
        for (int g = 0; g < 8; ++g) {
          int Mr = g + 8 * hi;
          float S = fmaxf(acc[g], 1e-30f);
          sm.f.alpha[Mr * 128 + n0 + l15] =
              sm.f.m[Mr] + MT + __logf(S) + sm.f.emit[Mr * 128 + n0 + l15];
        }
      }
      __syncthreads();
    }
    // logZ[b] = logsumexp_j(alpha + end), wave r -> row r
    {
      float4 av = *(const float4*)&sm.f.alpha[r * 128 + c4];
      float4 ev = *(const float4*)&end_[c4];
      float a0 = av.x + ev.x, a1 = av.y + ev.y, a2 = av.z + ev.z, a3 = av.w + ev.w;
      float mx = fmaxf(fmaxf(a0, a1), fmaxf(a2, a3));
      #pragma unroll
      for (int off = 16; off >= 1; off >>= 1) mx = fmaxf(mx, __shfl_xor(mx, off));
      float s = __expf(a0 - mx) + __expf(a1 - mx) + __expf(a2 - mx) + __expf(a3 - mx);
      #pragma unroll
      for (int off = 16; off >= 1; off >>= 1) s += __shfl_xor(s, off);
      if (lane == 0) logZ[b0 + r] = mx + __logf(s);
    }
  } else if (blk < 68) {
    // ----------------- Viterbi (max-plus + argmax) -----------------
    const int b = blk - 4;
    const int j = tid & 127, q = tid >> 7;     // q = quarter of the i-reduction
    float tr[32];                               // trans column j, rows q*32..q*32+31
    #pragma unroll
    for (int u = 0; u < 32; ++u) tr[u] = trans[(q * 32 + u) * 128 + j];
    float em = 0.f;
    if (q == 0) {
      sm.v.v[j] = start[j] + emissions[((size_t)b * T_) * C_ + j];
      em = emissions[((size_t)b * T_ + 1) * C_ + j];   // prefetch t=1
    }
    __syncthreads();

    for (int t = 1; t < T_; ++t) {
      float best = -3.4e38f; int bi = 0;
      #pragma unroll
      for (int k = 0; k < 8; ++k) {            // ds_load_b128 per 4 candidates
        float4 vv = *(const float4*)&sm.v.v[q * 32 + k * 4];
        float s;                               // strict > keeps first max (jnp.argmax)
        s = vv.x + tr[k*4+0]; if (s > best) { best = s; bi = k*4+0; }
        s = vv.y + tr[k*4+1]; if (s > best) { best = s; bi = k*4+1; }
        s = vv.z + tr[k*4+2]; if (s > best) { best = s; bi = k*4+2; }
        s = vv.w + tr[k*4+3]; if (s > best) { best = s; bi = k*4+3; }
      }
      PV mine; mine.v = best; mine.i = q * 32 + bi;
      sm.v.pvpi[q * 128 + j] = mine;
      __syncthreads();
      if (q == 0) {
        PV a = sm.v.pvpi[j];
        #pragma unroll
        for (int qq = 1; qq < 4; ++qq) {       // ascending q, strict > -> lowest index wins
          PV c = sm.v.pvpi[qq * 128 + j];
          if (c.v > a.v) a = c;
        }
        sm.v.v[j] = a.v + em;
        bp[(size_t)b * TM1 * C_ + (size_t)(t - 1) * C_ + j] = (unsigned char)a.i;
        if (t + 1 < T_)
          em = emissions[((size_t)b * T_ + t + 1) * C_ + j];   // prefetch next step
      }
      __syncthreads();
    }
    if (tid == 0) {
      float bb = -3.4e38f; int ii = 0;
      for (int jj = 0; jj < 128; ++jj) {
        float s = sm.v.v[jj] + end_[jj];
        if (s > bb) { bb = s; ii = jj; }
      }
      bestL[b] = ii;
    }
  } else {
    // ----------------- gold path score -----------------
    const int gb = blk - 68;
    const int rb = tid >> 6, l64 = tid & 63;
    const int b = gb * 8 + rb;
    float p = 0.f;
    for (int t = l64; t < T_; t += 64) {
      int tg = tags[(size_t)b * T_ + t];
      p += emissions[((size_t)b * T_ + t) * C_ + tg];
      if (t + 1 < T_) {
        int tg2 = tags[(size_t)b * T_ + t + 1];
        p += trans[tg * 128 + tg2];
      }
    }
    sm.g.red[tid] = p;
    __syncthreads();
    #pragma unroll
    for (int st = 32; st >= 1; st >>= 1) {
      if (l64 < st) sm.g.red[tid] += sm.g.red[tid + st];
      __syncthreads();
    }
    if (l64 == 0)
      gold[b] = sm.g.red[tid] + start[tags[(size_t)b * T_]]
                              + end_[tags[(size_t)b * T_ + TM1]];
  }
}

#define CHUNK 384   // bp timesteps staged per LDS chunk (48 KB)

__global__ __launch_bounds__(256)
void crf_backtrace(float* __restrict__ out, unsigned char* __restrict__ ws)
{
  __shared__ unsigned int  chunk[CHUNK * 32];   // 48 KB of backpointers
  __shared__ unsigned char path[T_];
  const int b = blockIdx.x, tid = threadIdx.x;
  const unsigned char* bpb = (const unsigned char*)out + 256 + (size_t)b * TM1 * C_;
  const int*   bestL = (const int*)(ws + WS_BEST);
  const float* logZ  = (const float*)(ws + WS_LOGZ);
  const float* gold  = (const float*)(ws + WS_GOLD);
  unsigned char* wpath = ws + WS_PATH + (size_t)b * T_;

  if (tid == 0) {
    out[b] = logZ[b] - gold[b];                 // nll
    path[T_ - 1] = (unsigned char)bestL[b];
  }
  __syncthreads();

  int s_hi = TM1 - 1;                           // bp index s = t-1, walk 2046..0
  while (s_hi >= 0) {
    int s_lo = s_hi - CHUNK + 1; if (s_lo < 0) s_lo = 0;
    int words = (s_hi - s_lo + 1) * 32;         // bytes/4
    const unsigned int* src = (const unsigned int*)(bpb + (size_t)s_lo * 128);
    for (int w = tid; w < words; w += 256) chunk[w] = src[w];
    __syncthreads();
    if (tid == 0) {
      const unsigned char* ch = (const unsigned char*)chunk;
      int cur = path[s_hi + 1];
      for (int s = s_hi; s >= s_lo; --s) {
        cur = ch[(s - s_lo) * 128 + cur];
        path[s] = (unsigned char)cur;
      }
    } else {
      // while lane 0 walks, warm GL2 with the next chunk (global_prefetch_b8)
      int n_hi = s_lo - 1;
      if (n_hi >= 0) {
        int n_lo = n_hi - CHUNK + 1; if (n_lo < 0) n_lo = 0;
        const unsigned char* nb = bpb + (size_t)n_lo * 128;
        size_t bytes = (size_t)(n_hi - n_lo + 1) * 128;
        for (size_t o = (size_t)(tid - 1) * 128; o < bytes; o += (size_t)255 * 128)
          __builtin_prefetch(nb + o, 0, 1);
      }
    }
    __syncthreads();
    s_hi = s_lo - 1;
  }
  for (int t = tid; t < T_; t += 256) wpath[t] = path[t];
}

__global__ __launch_bounds__(256)
void crf_onehot(float* __restrict__ out, const unsigned char* __restrict__ ws)
{
  const unsigned char* wpath = ws + WS_PATH;
  size_t idx4 = (size_t)blockIdx.x * 256 + threadIdx.x;  // one float4 each
  size_t f = idx4 * 4;                                   // flat onehot index
  int b   = (int)(f >> 18);                              // 2048*128 = 2^18
  int rem = (int)(f & 262143);
  int t = rem >> 7, c = rem & 127;
  int p = wpath[b * T_ + t];
  float4 v;
  v.x = (c     == p) ? 1.f : 0.f;
  v.y = (c + 1 == p) ? 1.f : 0.f;
  v.z = (c + 2 == p) ? 1.f : 0.f;
  v.w = (c + 3 == p) ? 1.f : 0.f;
  *(float4*)(out + 64 + f) = v;
}

extern "C" void kernel_launch(void* const* d_in, const int* in_sizes, int n_in,
                              void* d_out, int out_size, void* d_ws, size_t ws_size,
                              hipStream_t stream) {
  (void)in_sizes; (void)n_in; (void)out_size; (void)ws_size;
  const float* emissions = (const float*)d_in[0];
  const int*   tags      = (const int*)d_in[1];
  const float* trans     = (const float*)d_in[2];
  const float* start     = (const float*)d_in[3];
  const float* endt      = (const float*)d_in[4];
  float* out = (float*)d_out;
  unsigned char* ws = (unsigned char*)d_ws;

  crf_main<<<76, 512, 0, stream>>>(emissions, tags, trans, start, endt, out, ws);
  crf_backtrace<<<64, 256, 0, stream>>>(out, ws);
  // 64*2048*128/4 float4 elements = 4,194,304 -> 16384 blocks x 256 threads
  crf_onehot<<<16384, 256, 0, stream>>>(out, ws);
}
